// ICON_LM_738734375491
// MI455X (gfx1250) — matrix operand order repaired and checked by reading, not compiled, not run on hardware
//
#include <hip/hip_runtime.h>
#include <hip/hip_bf16.h>

typedef __bf16 bf16;
typedef __attribute__((ext_vector_type(16))) __bf16 v16bf;
typedef __attribute__((ext_vector_type(8)))  float  v8f;
typedef __attribute__((ext_vector_type(4)))  unsigned int u32x4;
typedef __attribute__((ext_vector_type(8)))  unsigned int u32x8;

#define Bsz   16
#define Lseq  750
#define Dm    1024
#define Hn    16
#define HDim  64
#define NLay  6
#define DFFn  4096
#define Mrows (Bsz*Lseq)     // 12000
#define QT    47             // ceil(750/16)
#define KCn   24             // 768/32 key chunks
#define LKP   768
#define LQP   752
#define NOUT  250            // DEMO*QOI
#define NEGF  -1000000000.0f

// ---------------- WMMA helpers (CDNA5 gfx1250, wave32) ----------------

static __device__ __forceinline__ v8f wmma_bf16(v16bf a, v16bf b, v8f c) {
  // v_wmma_f32_16x16x32_bf16  D = A(16x32) * B(32x16) + C
  return __builtin_amdgcn_wmma_f32_16x16x32_bf16(false, a, false, b, (short)0, c,
                                                 false, false);
}

// A-fragment (16x32 bf16). Caller passes per-lane pointer:
//   p = base + (lane&15)*row_stride + ((lane&16)?16:0)
// element e -> K = koff+e (p[e])
static __device__ __forceinline__ v16bf frag_ld_a(const bf16* __restrict__ p) {
  v16bf f;
#pragma unroll
  for (int i = 0; i < 8; ++i) f[i] = p[i];
#pragma unroll
  for (int i = 0; i < 8; ++i) f[i + 8] = p[i + 8];
  return f;
}

// B-fragment (32x16 bf16), K contiguous in memory. Caller passes
//   p = base + (lane&15)*n_stride + ((lane&16)?16:0);  element e -> K = koff+e
static __device__ __forceinline__ v16bf frag_ld_b(const bf16* __restrict__ p) {
  v16bf f;
#pragma unroll
  for (int i = 0; i < 16; ++i) f[i] = p[i];
  return f;
}

// B-fragment with strided K (for LDS-resident V: element e at p[e*stride])
static __device__ __forceinline__ v16bf frag_ld_b_str(const bf16* __restrict__ p,
                                                      int stride) {
  v16bf f;
#pragma unroll
  for (int i = 0; i < 16; ++i) f[i] = p[i * stride];
  return f;
}

static __device__ __forceinline__ float gelu_exact(float x) {
  return 0.5f * x * (1.0f + erff(x * 0.70710678118654752f));
}

// ---------------- prep kernels ----------------

// W (layers,K,N) fp32  ->  WT (layers,N,K) bf16
__global__ void wconv_kernel(const float* __restrict__ W, bf16* __restrict__ WT,
                             int K, int N, int layers) {
  size_t per = (size_t)K * N;
  size_t total = per * layers;
  for (size_t i = (size_t)blockIdx.x * blockDim.x + threadIdx.x; i < total;
       i += (size_t)gridDim.x * blockDim.x) {
    size_t l = i / per;
    size_t r = i - l * per;
    int n = (int)(r / K);
    int k = (int)(r - (size_t)n * K);
    WT[i] = (bf16)W[l * per + (size_t)k * N + n];
  }
}

// bool mask (750x750) -> padded f32 bias (752x768)
__global__ void maskbias_kernel(const unsigned char* __restrict__ mask,
                                float* __restrict__ biasPad) {
  int idx = blockIdx.x * blockDim.x + threadIdx.x;
  if (idx >= LQP * LKP) return;
  int q = idx / LKP, j = idx - q * LKP;
  float v;
  if (j >= Lseq)      v = NEGF;
  else if (q >= Lseq) v = (j == 0) ? 0.0f : NEGF;  // padded query rows: keep softmax finite
  else                v = mask[q * Lseq + j] ? 0.0f : NEGF;
  biasPad[idx] = v;
}

// x = seq @ pre_W + pre_b + pe[pos];  write fp32 + bf16 copies
__global__ void embed_kernel(const float* __restrict__ seq,
                             const float* __restrict__ preW,
                             const float* __restrict__ preb,
                             const float* __restrict__ pe,
                             const int* __restrict__ pos,
                             float* __restrict__ xf, bf16* __restrict__ xb) {
  size_t idx = (size_t)blockIdx.x * blockDim.x + threadIdx.x;
  if (idx >= (size_t)Mrows * Dm) return;
  int d = (int)(idx & (Dm - 1));
  size_t row = idx >> 10;
  int l = (int)(row % Lseq);
  float acc = preb[d] + pe[(size_t)pos[l] * Dm + d];
  const float* s = seq + row * 5;
#pragma unroll
  for (int i = 0; i < 5; ++i) acc += s[i] * preW[i * Dm + d];
  xf[idx] = acc;
  xb[idx] = (bf16)acc;
}

// ---------------- WMMA GEMM:  out = act(A(MxK) @ WT(NxK)^T + bias) ----------------
// block = 256 threads = 8 waves; waves 2(M)x4(N); wave tile 32x64; block tile 64x256

template <int ACT, bool WF32, bool WBF16>
__global__ __launch_bounds__(256) void gemm_wmma_kernel(
    const bf16* __restrict__ A, const bf16* __restrict__ WT,
    const float* __restrict__ bias, float* __restrict__ outF,
    bf16* __restrict__ outB, int M, int N, int K) {
  const int lane = threadIdx.x & 31;
  const int wave = threadIdx.x >> 5;
  const int wm = wave & 1;
  const int wn = wave >> 1;
  const int m0 = blockIdx.y * 64 + wm * 32;
  const int n0 = blockIdx.x * 256 + wn * 64;

  const int ar = lane & 15;
  const int akoff = (lane & 16) ? 16 : 0;
  const int bkoff = (lane & 16) ? 16 : 0;

  const int rowA0 = min(m0 + ar, M - 1);
  const int rowA1 = min(m0 + 16 + ar, M - 1);
  const bf16* pa0 = A + (size_t)rowA0 * K + akoff;
  const bf16* pa1 = A + (size_t)rowA1 * K + akoff;
  const bf16* pb[4];
#pragma unroll
  for (int nt = 0; nt < 4; ++nt)
    pb[nt] = WT + (size_t)(n0 + nt * 16 + ar) * K + bkoff;

  v8f acc[2][4];
#pragma unroll
  for (int mt = 0; mt < 2; ++mt)
#pragma unroll
    for (int nt = 0; nt < 4; ++nt)
      acc[mt][nt] = (v8f){0.f, 0.f, 0.f, 0.f, 0.f, 0.f, 0.f, 0.f};

  for (int kk = 0; kk < K; kk += 32) {
    __builtin_prefetch(pa0 + kk + 256, 0, 0);   // global_prefetch_b8
    __builtin_prefetch(pb[0] + kk + 256, 0, 0);
    v16bf a0 = frag_ld_a(pa0 + kk);
    v16bf a1 = frag_ld_a(pa1 + kk);
#pragma unroll
    for (int nt = 0; nt < 4; ++nt) {
      v16bf bfr = frag_ld_b(pb[nt] + kk);
      acc[0][nt] = wmma_bf16(a0, bfr, acc[0][nt]);
      acc[1][nt] = wmma_bf16(a1, bfr, acc[1][nt]);
    }
  }

  const int rr = (lane & 16) ? 8 : 0;
  const int col16 = lane & 15;
#pragma unroll
  for (int mt = 0; mt < 2; ++mt) {
#pragma unroll
    for (int nt = 0; nt < 4; ++nt) {
      const int colg = n0 + nt * 16 + col16;
      const float bb = bias[colg];
#pragma unroll
      for (int r = 0; r < 8; ++r) {
        int rowg = m0 + mt * 16 + rr + r;
        if (rowg < M) {
          float v = acc[mt][nt][r] + bb;
          if (ACT == 1) v = gelu_exact(v);
          if (WF32) outF[(size_t)rowg * N + colg] = v;
          if (WBF16) outB[(size_t)rowg * N + colg] = (bf16)v;
        }
      }
    }
  }
}

// ---------------- flash attention: one wave per (b, h, 16-query tile) ----------------
// V chunks are staged global -> LDS by the Tensor Data Mover (tensor_load_to_lds),
// overlapped with the QK^T WMMAs + softmax, synchronized with TENSORcnt.

__global__ __launch_bounds__(32) void attn_kernel(
    const bf16* __restrict__ qkv,      // (B*L, 3*D) bf16, [Q|K|V] each (H,HD)
    const float* __restrict__ biasPad, // (752, 768) f32
    bf16* __restrict__ outB) {         // (B*L, D) bf16
  __shared__ __align__(16) bf16 Plds[16 * 32];
  __shared__ __align__(16) bf16 Vlds[32 * HDim];

  const int lane = threadIdx.x & 31;
  int idx = blockIdx.x;
  const int qt = idx % QT;
  const int h = (idx / QT) & (Hn - 1);
  const int b = idx / (QT * Hn);
  const int q0 = qt * 16;

  const int ar = lane & 15;
  const int akoff = (lane & 16) ? 16 : 0;
  const int bkoff = (lane & 16) ? 16 : 0;
  const int rr = (lane & 16) ? 8 : 0;
  const int col16 = lane & 15;

  // Q fragments (hd 0..31, 32..63)
  const size_t rowQ = (size_t)b * Lseq + min(q0 + ar, Lseq - 1);
  const bf16* qbase = qkv + rowQ * (3 * Dm) + h * HDim + akoff;
  const v16bf qf0 = frag_ld_a(qbase);
  const v16bf qf1 = frag_ld_a(qbase + 32);

  // TDM descriptor pieces (uniform across the wave)
  const unsigned ldsAddrV = (unsigned)(uintptr_t)(void*)&Vlds[0];
  const unsigned long long gaV0 =
      (unsigned long long)(uintptr_t)(qkv + ((size_t)b * Lseq) * (3 * Dm) +
                                      2 * Dm + h * HDim);

  float m_run[8], l_run[8];
  v8f o[4];
#pragma unroll
  for (int r = 0; r < 8; ++r) { m_run[r] = -1e30f; l_run[r] = 0.f; }
#pragma unroll
  for (int nt = 0; nt < 4; ++nt)
    o[nt] = (v8f){0.f, 0.f, 0.f, 0.f, 0.f, 0.f, 0.f, 0.f};

  for (int kc = 0; kc < KCn; ++kc) {
    const int j0 = kc * 32;

    // WAR: previous chunk's PV ds-reads of Vlds must retire before TDM rewrites it
    asm volatile("s_wait_dscnt 0x0" ::: "memory");

    // ---- async V-chunk load (32 keys x 64 bf16 = 32x32 dwords) via TDM ----
    {
      const unsigned nrows = (unsigned)(Lseq - j0 < 32 ? Lseq - j0 : 32);
      const unsigned long long ga = gaV0 + (unsigned long long)j0 * (3 * Dm * 2);
      u32x4 d0;
      d0.x = 1u;                                       // count=1, user mode
      d0.y = ldsAddrV;                                 // LDS dest
      d0.z = (unsigned)ga;                             // global_addr[31:0]
      d0.w = (unsigned)(ga >> 32) | 0x80000000u;       // ga[56:32] | type=2
      u32x8 d1;
      d1.s0 = 2u << 16;        // data_size = 4 bytes
      d1.s1 = 32u << 16;       // tensor_dim0 = 32 dwords (one V row)
      d1.s2 = nrows << 16;     // tensor_dim1 = valid rows (tail reads as 0)
      d1.s3 = 32u << 16;       // tile_dim0 = 32 dwords
      d1.s4 = 32u;             // tile_dim1 = 32 rows, tile_dim2 = 0
      d1.s5 = 1536u;           // tensor_dim0_stride = 6144B / 4
      d1.s6 = 0u;
      d1.s7 = 0u;
      asm volatile("tensor_load_to_lds %0, %1" :: "s"(d0), "s"(d1) : "memory");
    }

    // ---- S = Q K^T over 32 keys (two 16-key tiles) ----
    v8f s[2];
#pragma unroll
    for (int t = 0; t < 2; ++t) {
      const size_t rowK = (size_t)b * Lseq + min(j0 + t * 16 + ar, Lseq - 1);
      const bf16* kb = qkv + rowK * (3 * Dm) + Dm + h * HDim + bkoff;
      v8f sv = (v8f){0.f, 0.f, 0.f, 0.f, 0.f, 0.f, 0.f, 0.f};
      sv = wmma_bf16(qf0, frag_ld_b(kb), sv);
      sv = wmma_bf16(qf1, frag_ld_b(kb + 32), sv);
      s[t] = sv;
    }

    // batched mask-bias loads (one clause, one wait)
    float bb0[8], bb1[8];
#pragma unroll
    for (int r = 0; r < 8; ++r) {
      const int q = q0 + rr + r;  // < 752
      bb0[r] = biasPad[q * LKP + j0 + col16];
      bb1[r] = biasPad[q * LKP + j0 + 16 + col16];
    }

    // scale + bias, chunk row max
    float cmax[8];
#pragma unroll
    for (int r = 0; r < 8; ++r) {
      s[0][r] = s[0][r] * 0.125f + bb0[r];
      s[1][r] = s[1][r] * 0.125f + bb1[r];
      cmax[r] = fmaxf(s[0][r], s[1][r]);
    }
#pragma unroll
    for (int d = 1; d < 16; d <<= 1)
#pragma unroll
      for (int r = 0; r < 8; ++r)
        cmax[r] = fmaxf(cmax[r], __shfl_xor(cmax[r], d, 32));

    // flash rescale
    float esc[8];
#pragma unroll
    for (int r = 0; r < 8; ++r) {
      float mn = fmaxf(m_run[r], cmax[r]);
      esc[r] = __expf(m_run[r] - mn);
      m_run[r] = mn;
      l_run[r] *= esc[r];
    }
#pragma unroll
    for (int nt = 0; nt < 4; ++nt)
#pragma unroll
      for (int r = 0; r < 8; ++r) o[nt][r] *= esc[r];

    // P = exp(S - m), row sums
    float psum[8];
#pragma unroll
    for (int r = 0; r < 8; ++r) {
      float p0 = __expf(s[0][r] - m_run[r]);
      float p1 = __expf(s[1][r] - m_run[r]);
      s[0][r] = p0; s[1][r] = p1;
      psum[r] = p0 + p1;
    }
#pragma unroll
    for (int d = 1; d < 16; d <<= 1)
#pragma unroll
      for (int r = 0; r < 8; ++r) psum[r] += __shfl_xor(psum[r], d, 32);
#pragma unroll
    for (int r = 0; r < 8; ++r) l_run[r] += psum[r];

    // P tile -> LDS (row-major 16x32 bf16)
#pragma unroll
    for (int t = 0; t < 2; ++t)
#pragma unroll
      for (int r = 0; r < 8; ++r)
        Plds[(rr + r) * 32 + t * 16 + col16] = (bf16)s[t][r];

    // P stores are same-wave in-order; V chunk must have landed via TDM
    asm volatile("s_wait_dscnt 0x0" ::: "memory");
    __builtin_amdgcn_s_wait_tensorcnt(0);

    // O += P(16x32) @ V(32x64)
    const v16bf pf = frag_ld_a(Plds + ar * 32 + akoff);
#pragma unroll
    for (int nt = 0; nt < 4; ++nt) {
      v16bf vf = frag_ld_b_str(Vlds + bkoff * HDim + nt * 16 + col16, HDim);
      o[nt] = wmma_bf16(pf, vf, o[nt]);
    }
  }

  // normalize + store
  float inv[8];
#pragma unroll
  for (int r = 0; r < 8; ++r) inv[r] = 1.0f / (l_run[r] + 1e-30f);
#pragma unroll
  for (int nt = 0; nt < 4; ++nt)
#pragma unroll
    for (int r = 0; r < 8; ++r) {
      int q = q0 + rr + r;
      if (q < Lseq)
        outB[((size_t)b * Lseq + q) * Dm + h * HDim + nt * 16 + col16] =
            (bf16)(o[nt][r] * inv[r]);
    }
}

// ---------------- residual + LayerNorm (one block per row) ----------------

__global__ __launch_bounds__(256) void add_ln_kernel(
    float* __restrict__ x, const float* __restrict__ delta,
    const float* __restrict__ g, const float* __restrict__ bt,
    bf16* __restrict__ xb) {
  const int tid = threadIdx.x;
  const size_t base = (size_t)blockIdx.x * Dm;
  float v[4];
  float s1 = 0.f, s2 = 0.f;
#pragma unroll
  for (int i = 0; i < 4; ++i) {
    float t = x[base + tid + i * 256] + delta[base + tid + i * 256];
    v[i] = t; s1 += t; s2 += t * t;
  }
#pragma unroll
  for (int d = 1; d < 32; d <<= 1) {
    s1 += __shfl_xor(s1, d, 32);
    s2 += __shfl_xor(s2, d, 32);
  }
  __shared__ float r1[8], r2[8];
  const int w = tid >> 5;
  if ((tid & 31) == 0) { r1[w] = s1; r2[w] = s2; }
  __syncthreads();
  if (w == 0) {
    float a = (tid < 8) ? r1[tid] : 0.f;
    float c = (tid < 8) ? r2[tid] : 0.f;
#pragma unroll
    for (int d = 1; d < 8; d <<= 1) {
      a += __shfl_xor(a, d, 32);
      c += __shfl_xor(c, d, 32);
    }
    if (tid == 0) { r1[0] = a; r2[0] = c; }
  }
  __syncthreads();
  const float mean = r1[0] * (1.0f / Dm);
  const float var = r2[0] * (1.0f / Dm) - mean * mean;
  const float rstd = rsqrtf(var + 1e-5f);
#pragma unroll
  for (int i = 0; i < 4; ++i) {
    int d = tid + i * 256;
    float ov = (v[i] - mean) * rstd * g[d] + bt[d];
    x[base + d] = ov;
    xb[base + d] = (bf16)ov;
  }
}

// ---------------- post projection + gather ----------------

__global__ __launch_bounds__(256) void post_kernel(
    const float* __restrict__ x, const float* __restrict__ pW,
    const float* __restrict__ pb, const int* __restrict__ oidx,
    float* __restrict__ y) {
  const int bo = blockIdx.x;           // b * NOUT + oi
  const int b = bo / NOUT;
  const int oi = bo - b * NOUT;
  const int l = oidx[oi];
  const float* xr = x + ((size_t)b * Lseq + l) * Dm;
  float a0 = 0.f, a1 = 0.f;
  for (int d = threadIdx.x; d < Dm; d += 256) {
    float xv = xr[d];
    a0 += xv * pW[d * 2 + 0];
    a1 += xv * pW[d * 2 + 1];
  }
#pragma unroll
  for (int d = 1; d < 32; d <<= 1) {
    a0 += __shfl_xor(a0, d, 32);
    a1 += __shfl_xor(a1, d, 32);
  }
  __shared__ float r0[8], r1[8];
  const int w = threadIdx.x >> 5;
  if ((threadIdx.x & 31) == 0) { r0[w] = a0; r1[w] = a1; }
  __syncthreads();
  if (threadIdx.x == 0) {
    float t0 = 0.f, t1 = 0.f;
#pragma unroll
    for (int i = 0; i < 8; ++i) { t0 += r0[i]; t1 += r1[i]; }
    y[bo * 2 + 0] = t0 + pb[0];
    y[bo * 2 + 1] = t1 + pb[1];
  }
}

// ---------------- launcher ----------------

extern "C" void kernel_launch(void* const* d_in, const int* in_sizes, int n_in,
                              void* d_out, int out_size, void* d_ws,
                              size_t ws_size, hipStream_t stream) {
  (void)in_sizes; (void)n_in; (void)out_size; (void)ws_size;
  const float* seq   = (const float*)d_in[0];
  const float* preW  = (const float*)d_in[1];
  const float* preb  = (const float*)d_in[2];
  const float* pe    = (const float*)d_in[3];
  const float* Wqkv  = (const float*)d_in[4];
  const float* bqkv  = (const float*)d_in[5];
  const float* Wo    = (const float*)d_in[6];
  const float* bo    = (const float*)d_in[7];
  const float* ln1g  = (const float*)d_in[8];
  const float* ln1b  = (const float*)d_in[9];
  const float* ln2g  = (const float*)d_in[10];
  const float* ln2b  = (const float*)d_in[11];
  const float* W1    = (const float*)d_in[12];
  const float* b1    = (const float*)d_in[13];
  const float* W2    = (const float*)d_in[14];
  const float* b2    = (const float*)d_in[15];
  const float* postW = (const float*)d_in[16];
  const float* postb = (const float*)d_in[17];
  const unsigned char* mask = (const unsigned char*)d_in[18];
  const int* ipos    = (const int*)d_in[19];
  const int* oidx    = (const int*)d_in[20];
  float* yout        = (float*)d_out;

  // workspace carve-up (256B aligned)
  char* ws = (char*)d_ws;
  auto carve = [&](size_t bytes) -> char* {
    char* p = ws;
    ws += (bytes + 255) & ~(size_t)255;
    return p;
  };
  bf16*  WqkvT = (bf16*)carve((size_t)NLay * 3 * Dm * Dm * sizeof(bf16));
  bf16*  WoT   = (bf16*)carve((size_t)NLay * Dm * Dm * sizeof(bf16));
  bf16*  W1T   = (bf16*)carve((size_t)NLay * DFFn * Dm * sizeof(bf16));
  bf16*  W2T   = (bf16*)carve((size_t)NLay * Dm * DFFn * sizeof(bf16));
  float* biasP = (float*)carve((size_t)LQP * LKP * sizeof(float));
  float* xf    = (float*)carve((size_t)Mrows * Dm * sizeof(float));
  bf16*  xb    = (bf16*)carve((size_t)Mrows * Dm * sizeof(bf16));
  bf16*  qkvb  = (bf16*)carve((size_t)Mrows * 3 * Dm * sizeof(bf16));
  bf16*  attb  = (bf16*)carve((size_t)Mrows * Dm * sizeof(bf16));
  bf16*  hb    = (bf16*)carve((size_t)Mrows * DFFn * sizeof(bf16));
  float* tmpF  = (float*)carve((size_t)Mrows * Dm * sizeof(float));

  // --- prep: bf16 transposed weights + padded mask bias ---
  wconv_kernel<<<4096, 256, 0, stream>>>(Wqkv, WqkvT, Dm, 3 * Dm, NLay);
  wconv_kernel<<<4096, 256, 0, stream>>>(Wo, WoT, Dm, Dm, NLay);
  wconv_kernel<<<4096, 256, 0, stream>>>(W1, W1T, Dm, DFFn, NLay);
  wconv_kernel<<<4096, 256, 0, stream>>>(W2, W2T, DFFn, Dm, NLay);
  maskbias_kernel<<<(LQP * LKP + 255) / 256, 256, 0, stream>>>(mask, biasP);

  // --- embed ---
  embed_kernel<<<(int)(((size_t)Mrows * Dm + 255) / 256), 256, 0, stream>>>(
      seq, preW, preb, pe, ipos, xf, xb);

  const dim3 gQKV(3 * Dm / 256, (Mrows + 63) / 64);
  const dim3 gD(Dm / 256, (Mrows + 63) / 64);
  const dim3 gFF(DFFn / 256, (Mrows + 63) / 64);

  for (int l = 0; l < NLay; ++l) {
    // QKV projection -> bf16
    gemm_wmma_kernel<0, false, true><<<gQKV, 256, 0, stream>>>(
        xb, WqkvT + (size_t)l * 3 * Dm * Dm, bqkv + (size_t)l * 3 * Dm,
        (float*)nullptr, qkvb, Mrows, 3 * Dm, Dm);
    // attention (TDM-staged V)
    attn_kernel<<<Bsz * Hn * QT, 32, 0, stream>>>(qkvb, biasP, attb);
    // output projection -> f32
    gemm_wmma_kernel<0, true, false><<<gD, 256, 0, stream>>>(
        attb, WoT + (size_t)l * Dm * Dm, bo + (size_t)l * Dm, tmpF,
        (bf16*)nullptr, Mrows, Dm, Dm);
    // x = LN(x + attn_proj)
    add_ln_kernel<<<Mrows, 256, 0, stream>>>(xf, tmpF, ln1g + (size_t)l * Dm,
                                             ln1b + (size_t)l * Dm, xb);
    // FFN up + exact GELU -> bf16
    gemm_wmma_kernel<1, false, true><<<gFF, 256, 0, stream>>>(
        xb, W1T + (size_t)l * DFFn * Dm, b1 + (size_t)l * DFFn,
        (float*)nullptr, hb, Mrows, DFFn, Dm);
    // FFN down -> f32
    gemm_wmma_kernel<0, true, false><<<gD, 256, 0, stream>>>(
        hb, W2T + (size_t)l * Dm * DFFn, b2 + (size_t)l * Dm, tmpF,
        (bf16*)nullptr, Mrows, Dm, DFFn);
    // x = LN(x + ffn)
    add_ln_kernel<<<Mrows, 256, 0, stream>>>(xf, tmpF, ln2g + (size_t)l * Dm,
                                             ln2b + (size_t)l * Dm, xb);
  }

  // post projection + gather
  post_kernel<<<Bsz * NOUT, 256, 0, stream>>>(xf, postW, postb, oidx, yout);
}
